// CrossAttention_21449066676378
// MI455X (gfx1250) — matrix operand order repaired and checked by reading, not compiled, not run on hardware
//
#include <hip/hip_runtime.h>
#include <hip/hip_bf16.h>
#include <math.h>

// ---------------------------------------------------------------------------
// CrossAttention fused pipeline for MI455X (gfx1250, wave32, WMMA).
// Precision: bf16 WMMA (v_wmma_f32_16x16x32_bf16) with fp32 accumulation.
// Flash-attention fusion avoids the 805MB attn matrix entirely.
// This revision hoists all B-fragment LDS loads ahead of each WMMA chain so
// the matrix ops issue back-to-back behind a single s_wait_dscnt.
// ---------------------------------------------------------------------------

typedef __bf16 bf16_t;
typedef __attribute__((ext_vector_type(16))) __bf16 v16bf;
typedef __attribute__((ext_vector_type(8)))  __bf16 v8bf;
typedef __attribute__((ext_vector_type(8)))  float  v8f;

#define B_  4
#define N_  2048
#define C_  768
#define H_  12
#define HD_ 64
#define M_  (B_ * N_)          // 8192 rows
#define EPS_ 1e-5f

// WMMA helper: D = A(16x32 bf16) * B(32x16 bf16) + C(16x16 f32)
__device__ __forceinline__ v8f wmma_bf16(v16bf a, v16bf b, v8f c) {
  return __builtin_amdgcn_wmma_f32_16x16x32_bf16(false, a, false, b,
                                                 (short)0, c, false, false);
}

// Build A-fragment from a row pointer in LDS/global:
// lane L: row = L%16, elems 0..15 -> K = (L/16)*16 + e
__device__ __forceinline__ v16bf make_afrag(const bf16_t* row, int half) {
  return *(const v16bf*)(row + half * 16);
}

// ---------------------------------------------------------------------------
// Kernel 1: projection GEMM (X[M,C] @ W[Dout,C]^T) with optional per-head
// LayerNorm epilogue; writes bf16 [B,H,N,HD]. One 64-col group == one head.
// Column groups < split go to outA (with lnA/scaleA), rest to outB.
// ---------------------------------------------------------------------------
__global__ __launch_bounds__(128)
void proj_ln_kernel(const float* __restrict__ X, const float* __restrict__ W,
                    bf16_t* __restrict__ outA, bf16_t* __restrict__ outB,
                    int split, int lnA, int lnB, float scaleA, float scaleB,
                    const float* __restrict__ lnw, const float* __restrict__ lnb)
{
  __shared__ __align__(32) bf16_t As[64][32];   // [row][k]
  __shared__ __align__(32) bf16_t Bs[64][32];   // [outcol][k]
  __shared__ float Cs[64][65];                  // epilogue staging

  const int t = threadIdx.x;
  const int wave = t >> 5, lane = t & 31;
  const int half = lane >> 4, l16 = lane & 15;
  const int row0 = blockIdx.x * 64;
  const int cg   = blockIdx.y;
  const int col0 = cg * 64;

  v8f acc[4] = {};

  for (int k0 = 0; k0 < C_; k0 += 32) {
    __syncthreads();
    // A tile: 64x32 f32 -> bf16 (each thread: 4 float4)
#pragma unroll
    for (int i = 0; i < 4; ++i) {
      int linear = t + i * 128;           // float4 id of 512
      int r = linear >> 3, kq = linear & 7;
      const float4 f = *(const float4*)(X + (size_t)(row0 + r) * C_ + k0 + kq * 4);
      bf16_t* dst = &As[r][kq * 4];
      dst[0] = (bf16_t)f.x; dst[1] = (bf16_t)f.y;
      dst[2] = (bf16_t)f.z; dst[3] = (bf16_t)f.w;
    }
    // B tile: W rows are K-contiguous (Linear uses x @ W^T)
#pragma unroll
    for (int i = 0; i < 4; ++i) {
      int linear = t + i * 128;
      int n = linear >> 3, kq = linear & 7;
      const float4 f = *(const float4*)(W + (size_t)(col0 + n) * C_ + k0 + kq * 4);
      bf16_t* dst = &Bs[n][kq * 4];
      dst[0] = (bf16_t)f.x; dst[1] = (bf16_t)f.y;
      dst[2] = (bf16_t)f.z; dst[3] = (bf16_t)f.w;
    }
    __syncthreads();

    // Load all fragments first, then issue the WMMA chain back-to-back.
    v16bf afrag = make_afrag(&As[wave * 16 + l16][0], half);
    v16bf bfr[4];
#pragma unroll
    for (int ct = 0; ct < 4; ++ct)
      bfr[ct] = *(const v16bf*)(&Bs[ct * 16 + l16][half * 16]);
#pragma unroll
    for (int ct = 0; ct < 4; ++ct)
      acc[ct] = wmma_bf16(afrag, bfr[ct], acc[ct]);
  }

  // dump accumulators: lane L elem r -> (row = r + 8*half, col = ct*16 + L%16)
#pragma unroll
  for (int ct = 0; ct < 4; ++ct)
#pragma unroll
    for (int r = 0; r < 8; ++r)
      Cs[wave * 16 + r + half * 8][ct * 16 + l16] = acc[ct][r];
  __syncthreads();

  const bool toA = (cg < split);
  bf16_t* outp = toA ? outA : outB;
  const int h = toA ? cg : (cg - split);
  const int doLN = toA ? lnA : lnB;
  const float scl = toA ? scaleA : scaleB;

  if (t < 64) {
    const int R = row0 + t;
    const int b = R / N_, n = R % N_;
    float mu = 0.f;
#pragma unroll
    for (int c = 0; c < 64; ++c) mu += Cs[t][c];
    mu *= (1.0f / 64.0f);
    float var = 0.f;
#pragma unroll
    for (int c = 0; c < 64; ++c) { float d = Cs[t][c] - mu; var += d * d; }
    var *= (1.0f / 64.0f);
    const float inv = rsqrtf(var + EPS_);
    bf16_t* orow = outp + ((size_t)(b * H_ + h) * N_ + n) * HD_;
#pragma unroll
    for (int c = 0; c < 64; ++c) {
      float x = Cs[t][c];
      float y = doLN ? ((x - mu) * inv * lnw[c] + lnb[c]) : x;
      orow[c] = (bf16_t)(y * scl);
    }
  }
}

// ---------------------------------------------------------------------------
// Kernel 2: flash attention. Block = (64 queries) x (b,h); 4 waves, each wave
// owns 16 query rows. Online softmax over 64-key tiles. Query-row mask folds
// into the running-max softmax (all-(-1e9) rows -> uniform weights, matching
// the reference semantics exactly).
// ---------------------------------------------------------------------------
__global__ __launch_bounds__(128)
void flash_attn_kernel(const bf16_t* __restrict__ Q,   // [B,H,N,HD]
                       const bf16_t* __restrict__ K,   // [B,H,N,HD]
                       const bf16_t* __restrict__ V,   // [B,H,N,HD]
                       const int* __restrict__ mask,   // [B,N]
                       bf16_t* __restrict__ O)         // [B,N,C]
{
  __shared__ __align__(32) bf16_t Ks[64][64];          // [key][d]
  __shared__ __align__(32) bf16_t VTs[64][64];         // [d][key]
  __shared__ __align__(32) bf16_t Ps[4][16][64];       // per-wave P staging

  const int t = threadIdx.x;
  const int wave = t >> 5, lane = t & 31;
  const int half = lane >> 4, l16 = lane & 15;
  const int qt = blockIdx.x, h = blockIdx.y, b = blockIdx.z;
  const size_t bh = (size_t)(b * H_ + h) * N_;
  const int q0 = qt * 64 + wave * 16;

  // Q fragments for this wave, kept in registers across the whole key loop.
  const bf16_t* qrow = Q + (bh + q0 + l16) * HD_;
  v16bf qfrag[2];
#pragma unroll
  for (int f = 0; f < 2; ++f) qfrag[f] = make_afrag(qrow + f * 32, half);

  // Row mask: elem r -> query row q0 + r + 8*half
  v8f mbias;
#pragma unroll
  for (int r = 0; r < 8; ++r)
    mbias[r] = (mask[b * N_ + q0 + r + half * 8] != 0) ? 0.0f : -1.0f;

  v8f oacc[4] = {};
  v8f mrun, lrun;
#pragma unroll
  for (int r = 0; r < 8; ++r) { mrun[r] = -3.0e38f; lrun[r] = 0.0f; }

  for (int kt = 0; kt < N_ / 64; ++kt) {
    __syncthreads();
    // K tile (row-major) and V tile (transposed) into LDS
#pragma unroll
    for (int i = 0; i < 4; ++i) {
      int linear = t + i * 128;          // v8bf id of 512
      int key = linear >> 3, dq = linear & 7;
      *(v8bf*)&Ks[key][dq * 8] =
          *(const v8bf*)(K + (bh + kt * 64 + key) * HD_ + dq * 8);
      v8bf vv = *(const v8bf*)(V + (bh + kt * 64 + key) * HD_ + dq * 8);
#pragma unroll
      for (int e = 0; e < 8; ++e) VTs[dq * 8 + e][key] = vv[e];
    }
    __syncthreads();

    if (kt + 1 < N_ / 64) {  // prefetch next tiles into caches
      __builtin_prefetch(K + (bh + (kt + 1) * 64 + (t & 63)) * HD_, 0, 0);
      __builtin_prefetch(V + (bh + (kt + 1) * 64 + (t & 63)) * HD_, 0, 0);
    }

    // S = Q * K^T  (16 x 64 per wave), K-dim = HD = 64 -> 2 wmma steps.
    // Load all 8 B-fragments, then run the 8-WMMA chain uninterrupted.
    v16bf kfr[2][4];
#pragma unroll
    for (int f = 0; f < 2; ++f)
#pragma unroll
      for (int ct = 0; ct < 4; ++ct)
        kfr[f][ct] = *(const v16bf*)(&Ks[ct * 16 + l16][f * 32 + half * 16]);

    v8f sacc[4] = {};
#pragma unroll
    for (int f = 0; f < 2; ++f)
#pragma unroll
      for (int ct = 0; ct < 4; ++ct)
        sacc[ct] = wmma_bf16(qfrag[f], kfr[f][ct], sacc[ct]);

    // Apply query-row mask (where(mask, s, -1e9))
#pragma unroll
    for (int ct = 0; ct < 4; ++ct)
#pragma unroll
      for (int r = 0; r < 8; ++r)
        if (mbias[r] < 0.0f) sacc[ct][r] = -1.0e9f;

    // Row max over 64 columns: elementwise over 4 col-tiles, then 16-lane xor
    v8f tmax = sacc[0];
#pragma unroll
    for (int ct = 1; ct < 4; ++ct)
#pragma unroll
      for (int r = 0; r < 8; ++r) tmax[r] = fmaxf(tmax[r], sacc[ct][r]);
    for (int mk = 1; mk < 16; mk <<= 1)
#pragma unroll
      for (int r = 0; r < 8; ++r)
        tmax[r] = fmaxf(tmax[r], __shfl_xor(tmax[r], mk, 32));

    v8f mnew, alpha;
#pragma unroll
    for (int r = 0; r < 8; ++r) {
      mnew[r]  = fmaxf(mrun[r], tmax[r]);
      alpha[r] = __expf(mrun[r] - mnew[r]);
    }

    // P = exp(S - m_new); row sums
    v8f rsum = {};
#pragma unroll
    for (int ct = 0; ct < 4; ++ct)
#pragma unroll
      for (int r = 0; r < 8; ++r) {
        float p = __expf(sacc[ct][r] - mnew[r]);
        sacc[ct][r] = p;
        rsum[r] += p;
      }
    for (int mk = 1; mk < 16; mk <<= 1)
#pragma unroll
      for (int r = 0; r < 8; ++r) rsum[r] += __shfl_xor(rsum[r], mk, 32);

#pragma unroll
    for (int r = 0; r < 8; ++r) {
      lrun[r] = lrun[r] * alpha[r] + rsum[r];
      mrun[r] = mnew[r];
    }
#pragma unroll
    for (int ct = 0; ct < 4; ++ct)
#pragma unroll
      for (int r = 0; r < 8; ++r) oacc[ct][r] *= alpha[r];

    // Stage P (C-layout) -> LDS so it can be re-read in A-fragment layout.
#pragma unroll
    for (int ct = 0; ct < 4; ++ct)
#pragma unroll
      for (int r = 0; r < 8; ++r)
        Ps[wave][r + half * 8][ct * 16 + l16] = (bf16_t)sacc[ct][r];
    asm volatile("s_wait_dscnt 0" ::: "memory");  // wave-local LDS RAW

    // O += P(16x64) * V(64x64): load P/V fragments, then 8-WMMA chain.
    v16bf pfr[2], vfr[2][4];
#pragma unroll
    for (int ks = 0; ks < 2; ++ks) {
      pfr[ks] = make_afrag(&Ps[wave][l16][ks * 32], half);
#pragma unroll
      for (int ct = 0; ct < 4; ++ct)
        vfr[ks][ct] = *(const v16bf*)(&VTs[ct * 16 + l16][ks * 32 + half * 16]);
    }
#pragma unroll
    for (int ks = 0; ks < 2; ++ks)
#pragma unroll
      for (int ct = 0; ct < 4; ++ct)
        oacc[ct] = wmma_bf16(pfr[ks], vfr[ks][ct], oacc[ct]);
  }

  // Finalize: O /= l, write bf16 [B,N,C] (head-interleaved) for the out-proj.
#pragma unroll
  for (int ct = 0; ct < 4; ++ct)
#pragma unroll
    for (int r = 0; r < 8; ++r) {
      const int q = q0 + r + half * 8;
      float o = oacc[ct][r] / lrun[r];
      O[((size_t)b * N_ + q) * C_ + h * HD_ + ct * 16 + l16] = (bf16_t)o;
    }
}

// ---------------------------------------------------------------------------
// Kernel 3: output projection. Xb[M,C] (bf16) @ Wp[C,C]^T + bp -> f32 out.
// ---------------------------------------------------------------------------
__global__ __launch_bounds__(128)
void out_proj_kernel(const bf16_t* __restrict__ Xb, const float* __restrict__ W,
                     const float* __restrict__ bias, float* __restrict__ out)
{
  __shared__ __align__(32) bf16_t As[64][32];
  __shared__ __align__(32) bf16_t Bs[64][32];

  const int t = threadIdx.x;
  const int wave = t >> 5, lane = t & 31;
  const int half = lane >> 4, l16 = lane & 15;
  const int row0 = blockIdx.x * 64;
  const int col0 = blockIdx.y * 64;

  v8f acc[4] = {};

  for (int k0 = 0; k0 < C_; k0 += 32) {
    __syncthreads();
    // A tile already bf16: 2048 elems = 256 v8bf chunks, 2 per thread
#pragma unroll
    for (int i = 0; i < 2; ++i) {
      int linear = t + i * 128;
      int r = linear >> 2, kq = linear & 3;
      *(v8bf*)&As[r][kq * 8] =
          *(const v8bf*)(Xb + (size_t)(row0 + r) * C_ + k0 + kq * 8);
    }
#pragma unroll
    for (int i = 0; i < 4; ++i) {
      int linear = t + i * 128;
      int n = linear >> 3, kq = linear & 7;
      const float4 f = *(const float4*)(W + (size_t)(col0 + n) * C_ + k0 + kq * 4);
      bf16_t* dst = &Bs[n][kq * 4];
      dst[0] = (bf16_t)f.x; dst[1] = (bf16_t)f.y;
      dst[2] = (bf16_t)f.z; dst[3] = (bf16_t)f.w;
    }
    __syncthreads();

    v16bf afrag = make_afrag(&As[wave * 16 + l16][0], half);
    v16bf bfr[4];
#pragma unroll
    for (int ct = 0; ct < 4; ++ct)
      bfr[ct] = *(const v16bf*)(&Bs[ct * 16 + l16][half * 16]);
#pragma unroll
    for (int ct = 0; ct < 4; ++ct)
      acc[ct] = wmma_bf16(afrag, bfr[ct], acc[ct]);
  }

#pragma unroll
  for (int ct = 0; ct < 4; ++ct) {
    const float bcol = bias[col0 + ct * 16 + l16];
#pragma unroll
    for (int r = 0; r < 8; ++r) {
      const int R = row0 + wave * 16 + r + half * 8;
      out[(size_t)R * C_ + col0 + ct * 16 + l16] = acc[ct][r] + bcol;
    }
  }
}

// ---------------------------------------------------------------------------
extern "C" void kernel_launch(void* const* d_in, const int* in_sizes, int n_in,
                              void* d_out, int out_size, void* d_ws, size_t ws_size,
                              hipStream_t stream) {
  const float* q_x  = (const float*)d_in[0];
  const float* kv_x = (const float*)d_in[1];
  const int*   mask = (const int*)d_in[2];
  const float* Wq   = (const float*)d_in[3];
  const float* Wkv  = (const float*)d_in[4];
  const float* qn_w = (const float*)d_in[5];
  const float* qn_b = (const float*)d_in[6];
  const float* kn_w = (const float*)d_in[7];
  const float* kn_b = (const float*)d_in[8];
  const float* Wp   = (const float*)d_in[9];
  const float* bp   = (const float*)d_in[10];
  float* out = (float*)d_out;

  // Workspace: Qb,Kb,Vb in [B,H,N,HD] bf16; Xb in [B,N,C] bf16. 4 x 12.6MB.
  const size_t seg = (size_t)B_ * H_ * N_ * HD_;   // elements
  bf16_t* Qb = (bf16_t*)d_ws;
  bf16_t* Kb = Qb + seg;
  bf16_t* Vb = Kb + seg;
  bf16_t* Xb = Vb + seg;

  const float qscale = 0.125f;  // HD^-0.5 = 64^-0.5

  dim3 blk(128);
  // Q = LN(q_x @ Wq^T) * scale        (12 head col-groups)
  proj_ln_kernel<<<dim3(M_ / 64, H_), blk, 0, stream>>>(
      q_x, Wq, Qb, Qb, H_, 1, 0, qscale, 1.0f, qn_w, qn_b);
  // [K;V] = kv_x @ Wkv^T: first 12 groups -> K (LN), last 12 -> V (raw)
  proj_ln_kernel<<<dim3(M_ / 64, 2 * H_), blk, 0, stream>>>(
      kv_x, Wkv, Kb, Vb, H_, 1, 0, 1.0f, 1.0f, kn_w, kn_b);
  // Flash attention -> Xb [B,N,C] bf16
  flash_attn_kernel<<<dim3(N_ / 64, H_, B_), blk, 0, stream>>>(
      Qb, Kb, Vb, mask, Xb);
  // out = Xb @ Wp^T + bp  (f32)
  out_proj_kernel<<<dim3(M_ / 64, H_), blk, 0, stream>>>(Xb, Wp, bp, out);
}
